// PEFTAdapter_34746285425328
// MI455X (gfx1250) — matrix operand order repaired and checked by reading, not compiled, not run on hardware
//
#include <hip/hip_runtime.h>
#include <math.h>

// ---------------------------------------------------------------------------
// PEFT adapter, fused: LayerNorm -> down(1536->64) -> GELU -> up(64->1536)
// -> +residual -> L2 normalize.  fp32 in/out, bf16 WMMA matmuls (fp32 acc).
// Target: MI455X gfx1250 (wave32, v_wmma_f32_16x16x32_bf16).
//
// Roofline: 12.9 GFLOP vs 403 MB unavoidable HBM traffic (read x + write out)
// => must use matrix cores (bf16 WMMA) to reach the ~17us bandwidth floor.
// Occupancy: __launch_bounds__(256,4) caps VGPRs at 256 so 2 workgroups
// (16 waves) fit per WGP; LDS is ~56KB/WG (2 fit in 320KB pool).
// ---------------------------------------------------------------------------

typedef __bf16 bf16;
typedef __attribute__((ext_vector_type(16))) __bf16 bf16x16;
typedef __attribute__((ext_vector_type(8)))  __bf16 bf16x8;
typedef __attribute__((ext_vector_type(4)))  __bf16 bf16x4;
typedef __attribute__((ext_vector_type(8)))  float  f32x8;

union ABf { bf16x16 v; bf16x8 h[2]; };

#define ED        1536     // embed dim
#define BN        64       // bottleneck
#define ROWS      16       // rows per workgroup (one 16xK WMMA tile in M)
#define XN_STRIDE 1544     // padded bf16 row stride (conflict-free b128 reads)
#define G_STRIDE  72       // padded bf16 stride for bottleneck activations
#define GP_STRIDE 68       // padded f32 stride for down-proj partials
#define NW        98304    // elements per weight matrix (64*1536)

__global__ void __launch_bounds__(256)
convert_weights_bf16(const float* __restrict__ wd,
                     const float* __restrict__ wu,
                     bf16* __restrict__ o)
{
    int t  = blockIdx.x * 256 + threadIdx.x;   // 49152 threads, 4 elems each
    int i4 = t * 4;
    const float* src = (i4 < NW) ? (wd + i4) : (wu + (i4 - NW));
    float4 v = *(const float4*)src;
    bf16x4 b;
    b[0] = (bf16)v.x; b[1] = (bf16)v.y; b[2] = (bf16)v.z; b[3] = (bf16)v.w;
    *(bf16x4*)(o + i4) = b;
}

__global__ void __launch_bounds__(256, 4)
peft_adapter_kernel(const float* __restrict__ x,
                    const bf16*  __restrict__ wd_bf,   // [64][1536] bf16
                    const float* __restrict__ b_down,  // [64]
                    const bf16*  __restrict__ wu_bf,   // [1536][64] bf16
                    const float* __restrict__ b_up,    // [1536]
                    const float* __restrict__ gamma,   // [1536]
                    const float* __restrict__ beta,    // [1536]
                    float* __restrict__ out)
{
    __shared__ __align__(16) bf16  xn[ROWS * XN_STRIDE];    // normalized x, bf16
    __shared__ __align__(16) bf16  gbuf[ROWS * G_STRIDE];   // GELU(down), bf16
    __shared__ __align__(16) float gpart[ROWS * GP_STRIDE]; // down K-split partials
    __shared__ float wsq[ROWS];
    __shared__ float rscale[ROWS];

    const int tid = threadIdx.x;
    if (tid < ROWS) wsq[tid] = 0.0f;

    // ---------------- P0: LayerNorm stats + normalize -> LDS (bf16) --------
    // 16 threads per row; thread covers d = seg*4 + i*64 (coalesced float4).
    const int row = tid >> 4;
    const int seg = tid & 15;
    const int rowg  = blockIdx.x * ROWS + row;
    const float* xrow = x + rowg * ED;

    float4 xa[24];
    float s1 = 0.0f, s2 = 0.0f;
    #pragma unroll
    for (int i = 0; i < 24; ++i) {
        xa[i] = *(const float4*)(xrow + seg * 4 + i * 64);
        s1 += xa[i].x + xa[i].y + xa[i].z + xa[i].w;
        s2 += xa[i].x * xa[i].x + xa[i].y * xa[i].y
            + xa[i].z * xa[i].z + xa[i].w * xa[i].w;
    }
    #pragma unroll
    for (int m = 8; m >= 1; m >>= 1) {
        s1 += __shfl_xor(s1, m, 16);
        s2 += __shfl_xor(s2, m, 16);
    }
    const float mean = s1 * (1.0f / ED);
    const float var  = s2 * (1.0f / ED) - mean * mean;
    const float rstd = rsqrtf(var + 1e-5f);

    #pragma unroll
    for (int i = 0; i < 24; ++i) {
        const int d = seg * 4 + i * 64;
        float4 g4 = *(const float4*)(gamma + d);
        float4 b4 = *(const float4*)(beta + d);
        bf16x4 o4;
        o4[0] = (bf16)((xa[i].x - mean) * rstd * g4.x + b4.x);
        o4[1] = (bf16)((xa[i].y - mean) * rstd * g4.y + b4.y);
        o4[2] = (bf16)((xa[i].z - mean) * rstd * g4.z + b4.z);
        o4[3] = (bf16)((xa[i].w - mean) * rstd * g4.w + b4.w);
        *(bf16x4*)(xn + row * XN_STRIDE + d) = o4;
    }
    __syncthreads();

    // ---------------- P1: down-proj, K split across wave pairs -------------
    const int wv   = tid >> 5;     // wave 0..7
    const int lane = tid & 31;
    const int half = lane >> 4;    // WMMA half-wave
    const int n16  = lane & 15;
    const int colb = (wv & 3) * 16;        // bottleneck tile base
    const int kbase = (wv >> 2) * 768;     // K half

    f32x8 acc;
    #pragma unroll
    for (int v = 0; v < 8; ++v) acc[v] = 0.0f;

    {
        const bf16* arow = xn + n16 * XN_STRIDE;            // A: row = lane&15
        const bf16* brow = wd_bf + (colb + n16) * ED;       // B: col = lane&15
        #pragma unroll 4
        for (int k0 = kbase; k0 < kbase + 768; k0 += 32) {
            ABf a, b;
            a.h[0] = *(const bf16x8*)(arow + k0 + 8 * half);
            a.h[1] = *(const bf16x8*)(arow + k0 + 16 + 8 * half);
            b.h[0] = *(const bf16x8*)(brow + k0 + 8 * half);
            b.h[1] = *(const bf16x8*)(brow + k0 + 16 + 8 * half);
            acc = __builtin_amdgcn_wmma_f32_16x16x32_bf16(
                false, a.v, false, b.v, (short)0, acc, false, false);
        }
    }

    if (wv >= 4) {  // high-K half stores partial
        #pragma unroll
        for (int v = 0; v < 8; ++v)
            gpart[(v + 8 * half) * GP_STRIDE + colb + n16] = acc[v];
    }
    __syncthreads();
    if (wv < 4) {   // combine halves + bias + exact GELU -> bf16 LDS
        const float bd = b_down[colb + n16];
        #pragma unroll
        for (int v = 0; v < 8; ++v) {
            float h = acc[v] + gpart[(v + 8 * half) * GP_STRIDE + colb + n16] + bd;
            float ge = 0.5f * h * (1.0f + erff(h * 0.70710678118654752f));
            gbuf[(v + 8 * half) * G_STRIDE + colb + n16] = (bf16)ge;
        }
    }
    __syncthreads();

    // ---------------- P2: up-proj (12 N-tiles per wave) + residual ---------
    ABf a0, a1;   // A operands (K=0..31, 32..63) reused across all N-tiles
    {
        const bf16* grow = gbuf + n16 * G_STRIDE;
        a0.h[0] = *(const bf16x8*)(grow + 0  + 8 * half);
        a0.h[1] = *(const bf16x8*)(grow + 16 + 8 * half);
        a1.h[0] = *(const bf16x8*)(grow + 32 + 8 * half);
        a1.h[1] = *(const bf16x8*)(grow + 48 + 8 * half);
    }

    f32x8 o12[12];
    float ss[8] = {0};
    const int xbase = blockIdx.x * ROWS * ED;

    #pragma unroll
    for (int i = 0; i < 12; ++i) {
        const int colg = (wv * 12 + i) * 16 + n16;   // 0..1535
        f32x8 c;
        const float bu = b_up[colg];
        #pragma unroll
        for (int v = 0; v < 8; ++v) c[v] = bu;       // C init = bias

        const bf16* burow = wu_bf + colg * BN;       // B: w_up row, contig in b
        ABf b0, b1;
        b0.h[0] = *(const bf16x8*)(burow + 8 * half);
        b0.h[1] = *(const bf16x8*)(burow + 16 + 8 * half);
        b1.h[0] = *(const bf16x8*)(burow + 32 + 8 * half);
        b1.h[1] = *(const bf16x8*)(burow + 48 + 8 * half);

        c = __builtin_amdgcn_wmma_f32_16x16x32_bf16(
            false, a0.v, false, b0.v, (short)0, c, false, false);
        c = __builtin_amdgcn_wmma_f32_16x16x32_bf16(
            false, a1.v, false, b1.v, (short)0, c, false, false);

        // residual add; last use of x -> non-temporal (free L2 early)
        #pragma unroll
        for (int v = 0; v < 8; ++v) {
            float xv = __builtin_nontemporal_load(
                x + xbase + (v + 8 * half) * ED + colg);
            float ov = xv + c[v];
            c[v] = ov;
            ss[v] += ov * ov;
        }
        o12[i] = c;
    }

    // ---------------- P3: row L2 norm across waves, scale, store -----------
    #pragma unroll
    for (int v = 0; v < 8; ++v) {
        #pragma unroll
        for (int m = 8; m >= 1; m >>= 1)
            ss[v] += __shfl_xor(ss[v], m, 16);
        if (n16 == 0)
            atomicAdd(&wsq[v + 8 * half], ss[v]);    // ds_add_f32
    }
    __syncthreads();
    if (tid < ROWS)
        rscale[tid] = 1.0f / fmaxf(sqrtf(wsq[tid]), 1e-12f);
    __syncthreads();

    float sc[8];
    #pragma unroll
    for (int v = 0; v < 8; ++v) sc[v] = rscale[v + 8 * half];

    #pragma unroll
    for (int i = 0; i < 12; ++i) {
        const int colg = (wv * 12 + i) * 16 + n16;
        #pragma unroll
        for (int v = 0; v < 8; ++v)
            __builtin_nontemporal_store(o12[i][v] * sc[v],
                out + xbase + (v + 8 * half) * ED + colg);  // streaming store
    }
}

// ---------------------------------------------------------------------------
extern "C" void kernel_launch(void* const* d_in, const int* in_sizes, int n_in,
                              void* d_out, int out_size, void* d_ws, size_t ws_size,
                              hipStream_t stream)
{
    const float* x      = (const float*)d_in[0];  // [32768,1536]
    const float* w_down = (const float*)d_in[1];  // [64,1536]
    const float* b_down = (const float*)d_in[2];  // [64]
    const float* w_up   = (const float*)d_in[3];  // [1536,64]
    const float* b_up   = (const float*)d_in[4];  // [1536]
    const float* gamma  = (const float*)d_in[5];  // [1536]
    const float* beta   = (const float*)d_in[6];  // [1536]
    float* outp = (float*)d_out;

    // Workspace: bf16 copies of both weight matrices (2 * 98304 * 2B = 384KB).
    bf16* wbf = (bf16*)d_ws;
    convert_weights_bf16<<<(2 * NW / 4 + 255) / 256, 256, 0, stream>>>(
        w_down, w_up, wbf);

    const int nrows = in_sizes[0] / ED;           // 32768
    peft_adapter_kernel<<<nrows / ROWS, 256, 0, stream>>>(
        x, wbf, b_down, wbf + NW, b_up, gamma, beta, outp);
}
